// Feature_extraction_layer_86526411145625
// MI455X (gfx1250) — matrix operand order repaired and checked by reading, not compiled, not run on hardware
//
#include <hip/hip_runtime.h>
#include <hip/hip_fp16.h>

typedef __attribute__((ext_vector_type(16))) _Float16 v16h;
typedef __attribute__((ext_vector_type(8)))  float    v8f;

#define B_    2
#define C_    5
#define P_    16384
#define S_    4096
#define K_    8
#define NCOL  (B_*S_*K_)   // 65536 attention columns
#define TPTS  2048         // LDS staging chunk for KNN

// ---------------------------------------------------------------- K1: min/max of x[:, :2, :] per (b,c)
__global__ void k_minmax(const float* __restrict__ x, float* __restrict__ mnmx) {
  __shared__ float smn[256], smx[256];
  int bc = blockIdx.x;            // b*2+c, 4 blocks
  int b = bc >> 1, c = bc & 1;
  const float* px = x + ((size_t)b * C_ + c) * P_;
  float mn = 1e30f, mx = -1e30f;
  for (int p = threadIdx.x; p < P_; p += 256) {
    float v = px[p];
    mn = fminf(mn, v); mx = fmaxf(mx, v);
  }
  smn[threadIdx.x] = mn; smx[threadIdx.x] = mx;
  __syncthreads();
  for (int s = 128; s > 0; s >>= 1) {
    if (threadIdx.x < s) {
      smn[threadIdx.x] = fminf(smn[threadIdx.x], smn[threadIdx.x + s]);
      smx[threadIdx.x] = fmaxf(smx[threadIdx.x], smx[threadIdx.x + s]);
    }
    __syncthreads();
  }
  if (threadIdx.x == 0) { mnmx[bc] = smn[0]; mnmx[4 + bc] = smx[0]; }
}

// ---------------------------------------------------------------- K2: 2D KNN-1 from mesh samples -> sp (first 3 channels)
// 8 waves/block share one batch's point cloud via LDS chunks (16 KB).
__global__ void __launch_bounds__(256) k_knn1(const float* __restrict__ x,
                                              const float* __restrict__ mnmx,
                                              float* __restrict__ sp3) {
  __shared__ float sx[TPTS], sy[TPTS];
  int wid  = (blockIdx.x * 256 + threadIdx.x) >> 5;   // 8192 waves = (b, ss); 8|4096 -> block shares b
  int lane = threadIdx.x & 31;
  int b = wid >> 12, ss = wid & (S_ - 1);
  float mn0 = mnmx[b*2+0], mn1 = mnmx[b*2+1];
  float mx0 = mnmx[4+b*2+0], mx1 = mnmx[4+b*2+1];
  // mesh: m01[j]=(j%64)/64 on channel0, m02[j]=(j/64)/64 on channel1
  float qx = ((float)(ss & 63) * (1.0f/64.0f)) * (mx0 - mn0) + mn0;
  float qy = ((float)(ss >> 6) * (1.0f/64.0f)) * (mx1 - mn1) + mn1;
  const float* px = x + ((size_t)b * C_ + 0) * P_;
  const float* py = x + ((size_t)b * C_ + 1) * P_;
  float bd = 1e30f; int bp = 0x7FFFFFFF;
  for (int chunk = 0; chunk < P_; chunk += TPTS) {
    __syncthreads();
    for (int t = threadIdx.x; t < TPTS; t += 256) {
      sx[t] = px[chunk + t]; sy[t] = py[chunk + t];
    }
    __syncthreads();
    for (int t = lane; t < TPTS; t += 32) {
      float dx = sx[t] - qx, dy = sy[t] - qy;
      float d = dx*dx + dy*dy;
      if (d < bd) { bd = d; bp = chunk + t; }    // strict <, increasing p: lowest index on ties
    }
  }
  for (int m = 16; m; m >>= 1) {                 // lexicographic (dist, idx) min
    float od = __shfl_xor(bd, m, 32);
    int   op = __shfl_xor(bp, m, 32);
    if (od < bd || (od == bd && op < bp)) { bd = od; bp = op; }
  }
  if (lane < 3)
    sp3[((size_t)b*3 + lane) * S_ + ss] = x[((size_t)b*C_ + lane) * P_ + bp];
}

// ---------------------------------------------------------------- K3: 3D KNN top-8 -> nbr[b][ss][0..7] (LDS-staged, 24 KB)
__global__ void __launch_bounds__(256) k_knn2(const float* __restrict__ x,
                                              const float* __restrict__ sp3,
                                              int* __restrict__ nbr) {
  __shared__ float sx[TPTS], sy[TPTS], sz[TPTS];
  int wid  = (blockIdx.x * 256 + threadIdx.x) >> 5;   // 8192 waves, block shares b
  int lane = threadIdx.x & 31;
  int b = wid >> 12, ss = wid & (S_ - 1);
  float qx = sp3[((size_t)b*3 + 0) * S_ + ss];
  float qy = sp3[((size_t)b*3 + 1) * S_ + ss];
  float qz = sp3[((size_t)b*3 + 2) * S_ + ss];
  const float* px = x + ((size_t)b * C_ + 0) * P_;
  const float* py = x + ((size_t)b * C_ + 1) * P_;
  const float* pz = x + ((size_t)b * C_ + 2) * P_;
  float bd[8]; int bi[8];
  #pragma unroll
  for (int i = 0; i < 8; ++i) { bd[i] = 1e30f; bi[i] = 0x7FFFFFFF; }
  for (int chunk = 0; chunk < P_; chunk += TPTS) {
    __syncthreads();
    for (int t = threadIdx.x; t < TPTS; t += 256) {
      __builtin_prefetch(px + chunk + TPTS + t, 0, 1);   // global_prefetch_b8 for next chunk
      sx[t] = px[chunk + t]; sy[t] = py[chunk + t]; sz[t] = pz[chunk + t];
    }
    __syncthreads();
    for (int t = lane; t < TPTS; t += 32) {
      float dx = sx[t]-qx, dy = sy[t]-qy, dz = sz[t]-qz;
      float d = dx*dx + dy*dy + dz*dz;
      if (d < bd[7]) {
        float nd = d; int ni = chunk + t;       // register-resident bubble insert (static indices)
        #pragma unroll
        for (int q = 0; q < 8; ++q) {
          if (nd < bd[q]) { float td = bd[q]; int ti = bi[q]; bd[q] = nd; bi[q] = ni; nd = td; ni = ti; }
        }
      }
    }
  }
  // wave-wide 8-round merge: pop global (dist, idx)-lex min each round
  int wlist[8];
  for (int r = 0; r < 8; ++r) {
    float cd = bd[0]; int ci = bi[0];
    float rd = cd;    int ri = ci;
    for (int m = 16; m; m >>= 1) {
      float od = __shfl_xor(rd, m, 32);
      int   oi = __shfl_xor(ri, m, 32);
      if (od < rd || (od == rd && oi < ri)) { rd = od; ri = oi; }
    }
    wlist[r] = ri;
    if (ci == ri) {                          // this lane's head consumed: shift down
      #pragma unroll
      for (int q = 0; q < 7; ++q) { bd[q] = bd[q+1]; bi[q] = bi[q+1]; }
      bd[7] = 1e30f; bi[7] = 0x7FFFFFFF;
    }
  }
  if (lane < 8) nbr[((size_t)b * S_ + ss) * 8 + lane] = wlist[lane];
}

// ---------------------------------------------------------------- K3b: scrambled gather (replicates _gather_nb reshape)
__global__ void k_gather(const float* __restrict__ x, const int* __restrict__ nbr,
                         float* __restrict__ x1) {
  int t = blockIdx.x * 256 + threadIdx.x;    // 65536
  int b = t >> 15, r = t & 32767;
  int i = r >> 3, jj = r & 7;                // output (i, jj)
  int kk = r >> 12, ssi = r & 4095;          // source (ss, kk): t' = i*8+jj = r
  int p = nbr[((size_t)b * S_ + ssi) * 8 + kk];
  #pragma unroll
  for (int c = 0; c < 5; ++c)
    x1[(((size_t)b*C_ + c) * S_ + i) * K_ + jj] = x[((size_t)b*C_ + c) * P_ + p];
}

// ---------------------------------------------------------------- K4: position-embed conv (10->32) + conv (4->32), pre-BN
__global__ void k_conv(const float* __restrict__ x1,
                       const float* __restrict__ Wpos,  const float* __restrict__ bpos,
                       const float* __restrict__ Wconv, const float* __restrict__ bconv,
                       float* __restrict__ pemb_raw, float* __restrict__ xc_raw) {
  int t = blockIdx.x * 256 + threadIdx.x;    // 65536
  int b = t >> 15, r = t & 32767;
  int i = r >> 3, kk = r & 7;
  float in10[10], in4[4];
  #pragma unroll
  for (int c = 0; c < 3; ++c) {
    float pa = x1[(((size_t)b*C_ + c) * S_ + i) * 8 + kk];
    float pc = x1[(((size_t)b*C_ + c) * S_ + i) * 8 + 0];
    in10[c] = pa; in10[3+c] = pc; in10[6+c] = pa - pc;
  }
  in10[9] = sqrtf(in10[6]*in10[6] + in10[7]*in10[7] + in10[8]*in10[8]);
  in4[0] = in10[0]; in4[1] = in10[1]; in4[2] = in10[2];
  in4[3] = x1[(((size_t)b*C_ + 3) * S_ + i) * 8 + kk];
  size_t obase = (size_t)b * 32 * 32768 + (size_t)i * 8 + kk;
  for (int o = 0; o < 32; ++o) {
    float a = bpos[o];
    #pragma unroll
    for (int c = 0; c < 10; ++c) a += Wpos[o*10 + c] * in10[c];
    pemb_raw[obase + (size_t)o * 32768] = a;
    float e = bconv[o];
    #pragma unroll
    for (int c = 0; c < 4; ++c) e += Wconv[o*4 + c] * in4[c];
    xc_raw[obase + (size_t)o * 32768] = e;
  }
}

// ---------------------------------------------------------------- K5: BN statistics per channel (ch<32: xc/g1, ch>=32: pemb/g2)
__global__ void k_bnstat(const float* __restrict__ xc_raw, const float* __restrict__ pemb_raw,
                         const float* __restrict__ g1, const float* __restrict__ be1,
                         const float* __restrict__ g2, const float* __restrict__ be2,
                         float* __restrict__ bnsc, float* __restrict__ bnsh) {
  __shared__ float s1[256], s2[256];
  int ch = blockIdx.x;                       // 64 channels
  const float* src = (ch < 32) ? xc_raw : pemb_raw;
  int c = ch & 31;
  float sum = 0.f, sq = 0.f;
  for (int e = threadIdx.x; e < NCOL; e += 256) {
    int b = e >> 15, rest = e & 32767;
    float v = src[((size_t)b * 32 + c) * 32768 + rest];
    sum += v; sq += v * v;
  }
  s1[threadIdx.x] = sum; s2[threadIdx.x] = sq;
  __syncthreads();
  for (int s = 128; s > 0; s >>= 1) {
    if (threadIdx.x < s) { s1[threadIdx.x] += s1[threadIdx.x+s]; s2[threadIdx.x] += s2[threadIdx.x+s]; }
    __syncthreads();
  }
  if (threadIdx.x == 0) {
    float mean = s1[0] * (1.0f / NCOL);
    float var  = s2[0] * (1.0f / NCOL) - mean * mean;   // biased, matches reference
    float g  = (ch < 32) ? g1[c]  : g2[c];
    float be = (ch < 32) ? be1[c] : be2[c];
    float sc = g * rsqrtf(var + 1e-5f);
    bnsc[ch] = sc; bnsh[ch] = be - mean * sc;
  }
}

// ---------------------------------------------------------------- K6: BN + lrelu(0.2), pack feat column-major f16 (WMMA B layout)
__global__ void k_feat(const float* __restrict__ xc_raw, const float* __restrict__ pemb_raw,
                       const float* __restrict__ bnsc, const float* __restrict__ bnsh,
                       _Float16* __restrict__ feath) {
  int n = blockIdx.x * 256 + threadIdx.x;    // 65536 columns: n = ((b*4096+j)*8+kk)
  int b = n >> 15, rest = n & 32767;
  _Float16* dst = feath + (size_t)n * 64;
  for (int c = 0; c < 32; ++c) {
    float v = xc_raw  [((size_t)b*32 + c) * 32768 + rest] * bnsc[c]    + bnsh[c];
    v = (v >= 0.f) ? v : 0.2f * v;
    dst[c] = (_Float16)v;
    float w = pemb_raw[((size_t)b*32 + c) * 32768 + rest] * bnsc[32+c] + bnsh[32+c];
    w = (w >= 0.f) ? w : 0.2f * w;
    dst[32 + c] = (_Float16)w;
  }
}

// ---------------------------------------------------------------- K_att: fused WMMA attention GEMM + softmax(k) + weighted sum
// One wave per 16-column tile (= 2 sample points x 8 neighbors). 4 x v_wmma_f32_16x16x32_f16.
// Emits pooled features in f16 column-major (WMMA B layout for the next GEMM).
__global__ void __launch_bounds__(256) k_att(const float* __restrict__ Wa1, const float* __restrict__ ba1,
                                             const float* __restrict__ Wa2, const float* __restrict__ ba2,
                                             const _Float16* __restrict__ feath,
                                             _Float16* __restrict__ fsh) {
  int wid  = (blockIdx.x * 256 + threadIdx.x) >> 5;   // 4096 tiles, exact grid -> EXEC all ones
  int lane = threadIdx.x & 31;
  int g = lane >> 4, m = lane & 15;
  int nbase = wid << 4;
  int b  = nbase >> 15;
  int j0 = (nbase >> 3) & 4095;

  // A operands: W_a1 rows [0..15] (A0*) and [16..31] (A1*); contiguous k = 16*g + e layout (matches B).
  v16h A0a, A0b, A1a, A1b;
  #pragma unroll
  for (int v = 0; v < 8; ++v) {
    #pragma unroll
    for (int jh = 0; jh < 2; ++jh) {
      int Kk = (g << 4) + (v << 1) + jh;  // K index 0..31
      int e  = (v << 1) + jh;
      A0a[e] = (_Float16)Wa1[m * 64 + Kk];
      A0b[e] = (_Float16)Wa1[m * 64 + 32 + Kk];
      A1a[e] = (_Float16)Wa1[(m + 16) * 64 + Kk];
      A1b[e] = (_Float16)Wa1[(m + 16) * 64 + 32 + Kk];
    }
  }
  // B operands: lane = column (n = lane%16), element e = K row g*16+e -> contiguous channels.
  const _Float16* frow = feath + (size_t)(nbase + m) * 64;
  v16h Bv0 = *(const v16h*)(frow + g * 16);        // channels [g*16, g*16+16)
  v16h Bv1 = *(const v16h*)(frow + 32 + g * 16);   // channels [32+g*16, ...)

  v8f C0 = {0.f,0.f,0.f,0.f,0.f,0.f,0.f,0.f};
  v8f C1 = {0.f,0.f,0.f,0.f,0.f,0.f,0.f,0.f};
  C0 = __builtin_amdgcn_wmma_f32_16x16x32_f16(false, A0a, false, Bv0, (short)0, C0, false, false);
  C0 = __builtin_amdgcn_wmma_f32_16x16x32_f16(false, A0b, false, Bv1, (short)0, C0, false, false);
  C1 = __builtin_amdgcn_wmma_f32_16x16x32_f16(false, A1a, false, Bv0, (short)0, C1, false, false);
  C1 = __builtin_amdgcn_wmma_f32_16x16x32_f16(false, A1b, false, Bv1, (short)0, C1, false, false);

  // bias + lrelu(0.2) + W_a2 dot (partial per lane; rows M = r + 8g and 16+M)
  float p = 0.f;
  #pragma unroll
  for (int r = 0; r < 8; ++r) {
    int M0 = r + (g << 3);
    float h0 = C0[r] + ba1[M0];      h0 = (h0 >= 0.f) ? h0 : 0.2f * h0;
    float h1 = C1[r] + ba1[16 + M0]; h1 = (h1 >= 0.f) ? h1 : 0.2f * h1;
    p += Wa2[M0] * h0 + Wa2[16 + M0] * h1;
  }
  p += __shfl_xor(p, 16, 32);                 // combine the two 8-row halves of a column
  float logit = p + ba2[0];

  // softmax over k=8 columns (lanes differing in bits 0..2 share a sample point)
  float mx = logit;
  mx = fmaxf(mx, __shfl_xor(mx, 1, 32));
  mx = fmaxf(mx, __shfl_xor(mx, 2, 32));
  mx = fmaxf(mx, __shfl_xor(mx, 4, 32));
  float e1 = __expf(logit - mx);
  float sm = e1;
  sm += __shfl_xor(sm, 1, 32);
  sm += __shfl_xor(sm, 2, 32);
  sm += __shfl_xor(sm, 4, 32);
  float att = e1 / sm;

  // weighted feature sum over k, reusing B operands already in registers
  float acc[32];
  #pragma unroll
  for (int i = 0; i < 16; ++i) { acc[i] = (float)Bv0[i] * att; acc[16 + i] = (float)Bv1[i] * att; }
  #pragma unroll
  for (int msk = 1; msk <= 4; msk <<= 1) {
    #pragma unroll
    for (int i = 0; i < 32; ++i) acc[i] += __shfl_xor(acc[i], msk, 32);
  }
  if ((lane & 7) == 0) {                      // lanes 0,8,16,24 hold the reduced sums
    int col = b * S_ + j0 + ((lane >> 3) & 1);
    v16h o0, o1;
    #pragma unroll
    for (int i = 0; i < 16; ++i) { o0[i] = (_Float16)acc[i]; o1[i] = (_Float16)acc[16 + i]; }
    _Float16* dst = fsh + (size_t)col * 64;
    *(v16h*)(dst + g * 16)      = o0;         // channels [g*16, g*16+16)
    *(v16h*)(dst + 32 + g * 16) = o1;         // channels [32+g*16, ...)
  }
}

// ---------------------------------------------------------------- K_fea1: Fea1 = W_b1(64x64) x fsh(64x8192) via WMMA
// One wave per 16-column tile: 4 row tiles x 2 K-steps = 8 x v_wmma_f32_16x16x32_f16.
__global__ void __launch_bounds__(256) k_fea1(const float* __restrict__ Wb1,
                                              const _Float16* __restrict__ fsh,
                                              float* __restrict__ fea1) {
  int wid  = (blockIdx.x * 256 + threadIdx.x) >> 5;   // 512 tiles over 8192 cols, exact grid
  int lane = threadIdx.x & 31;
  int g = lane >> 4, m = lane & 15;
  int col0 = wid << 4;

  v16h A[4][2];                               // 4 row tiles x {K 0..31, K 32..63}
  #pragma unroll
  for (int rt = 0; rt < 4; ++rt) {
    #pragma unroll
    for (int v = 0; v < 8; ++v) {
      #pragma unroll
      for (int jh = 0; jh < 2; ++jh) {
        int Kk = (g << 4) + (v << 1) + jh;
        int e  = (v << 1) + jh;
        A[rt][0][e] = (_Float16)Wb1[(rt * 16 + m) * 64 + Kk];
        A[rt][1][e] = (_Float16)Wb1[(rt * 16 + m) * 64 + 32 + Kk];
      }
    }
  }
  const _Float16* frow = fsh + (size_t)(col0 + m) * 64;
  v16h B0 = *(const v16h*)(frow + g * 16);
  v16h B1 = *(const v16h*)(frow + 32 + g * 16);

  #pragma unroll
  for (int rt = 0; rt < 4; ++rt) {
    v8f C = {0.f,0.f,0.f,0.f,0.f,0.f,0.f,0.f};
    C = __builtin_amdgcn_wmma_f32_16x16x32_f16(false, A[rt][0], false, B0, (short)0, C, false, false);
    C = __builtin_amdgcn_wmma_f32_16x16x32_f16(false, A[rt][1], false, B1, (short)0, C, false, false);
    // D: lane holds column col0+m, rows rt*16 + 8g + r  -> contiguous in col-major fea1[col][row]
    float* dst = fea1 + (size_t)(col0 + m) * 64 + rt * 16 + 8 * g;
    #pragma unroll
    for (int r = 0; r < 8; ++r) dst[r] = C[r];
  }
}

// ---------------------------------------------------------------- K7: epilogue: lrelu(Fea1 + Fea2, 0.01), concat x1[:, :5, :, 0]
__global__ void k_out(const float* __restrict__ x1, const float* __restrict__ fea1,
                      const float* __restrict__ bb1,
                      const float* __restrict__ Wb2, const float* __restrict__ bb2,
                      float* __restrict__ out) {
  int t = blockIdx.x * 256 + threadIdx.x;
  if (t >= B_ * 69 * S_) return;
  int b = t / (69 * S_), r = t % (69 * S_);
  int ch = r / S_, j = r % S_;
  float val;
  if (ch < 5) {
    val = x1[(((size_t)b*C_ + ch) * S_ + j) * 8];
  } else {
    int o = ch - 5;
    float a = fea1[((size_t)(b * S_ + j)) * 64 + o] + bb1[o] + bb2[o];
    #pragma unroll
    for (int c = 0; c < 4; ++c)  a += Wb2[o*4 + c] * x1[(((size_t)b*C_ + c) * S_ + j) * 8];
    val = (a >= 0.f) ? a : 0.01f * a;
  }
  out[((size_t)b * 69 + ch) * S_ + j] = val;
}

// ---------------------------------------------------------------- launch
extern "C" void kernel_launch(void* const* d_in, const int* in_sizes, int n_in,
                              void* d_out, int out_size, void* d_ws, size_t ws_size,
                              hipStream_t stream) {
  const float* x     = (const float*)d_in[0];
  const float* Wpos  = (const float*)d_in[1];
  const float* bpos  = (const float*)d_in[2];
  const float* g2    = (const float*)d_in[3];
  const float* be2   = (const float*)d_in[4];
  const float* Wconv = (const float*)d_in[5];
  const float* bconv = (const float*)d_in[6];
  const float* g1    = (const float*)d_in[7];
  const float* be1   = (const float*)d_in[8];
  const float* Wa1   = (const float*)d_in[9];
  const float* ba1   = (const float*)d_in[10];
  const float* Wa2   = (const float*)d_in[11];
  const float* ba2   = (const float*)d_in[12];
  const float* Wb1   = (const float*)d_in[13];
  const float* bb1   = (const float*)d_in[14];
  const float* Wb2   = (const float*)d_in[15];
  const float* bb2   = (const float*)d_in[16];
  float* out = (float*)d_out;

  char* ws = (char*)d_ws;
  size_t off = 0;
  auto take = [&](size_t bytes) { size_t o = off; off = (off + bytes + 255) & ~(size_t)255; return o; };
  float*    mnmx    = (float*)   (ws + take(8 * 4));
  float*    sp3     = (float*)   (ws + take((size_t)B_ * 3 * S_ * 4));
  int*      nbr     = (int*)     (ws + take((size_t)B_ * S_ * 8 * 4));
  float*    x1      = (float*)   (ws + take((size_t)B_ * C_ * S_ * K_ * 4));
  float*    pembraw = (float*)   (ws + take((size_t)B_ * 32 * S_ * K_ * 4));
  float*    xcraw   = (float*)   (ws + take((size_t)B_ * 32 * S_ * K_ * 4));
  float*    bnsc    = (float*)   (ws + take(64 * 4));
  float*    bnsh    = (float*)   (ws + take(64 * 4));
  _Float16* feath   = (_Float16*)(ws + take((size_t)NCOL * 64 * 2));
  _Float16* fsh     = (_Float16*)(ws + take((size_t)B_ * S_ * 64 * 2));
  float*    fea1    = (float*)   (ws + take((size_t)B_ * S_ * 64 * 4));

  k_minmax<<<4,    256, 0, stream>>>(x, mnmx);
  k_knn1  <<<1024, 256, 0, stream>>>(x, mnmx, sp3);              // 8192 waves exactly
  k_knn2  <<<1024, 256, 0, stream>>>(x, sp3, nbr);               // 8192 waves exactly
  k_gather<<<256,  256, 0, stream>>>(x, nbr, x1);
  k_conv  <<<256,  256, 0, stream>>>(x1, Wpos, bpos, Wconv, bconv, pembraw, xcraw);
  k_bnstat<<<64,   256, 0, stream>>>(xcraw, pembraw, g1, be1, g2, be2, bnsc, bnsh);
  k_feat  <<<256,  256, 0, stream>>>(xcraw, pembraw, bnsc, bnsh, feath);
  k_att   <<<512,  256, 0, stream>>>(Wa1, ba1, Wa2, ba2, feath, fsh);   // 4096 WMMA tiles exactly
  k_fea1  <<<64,   256, 0, stream>>>(Wb1, fsh, fea1);                   // 512 WMMA tiles exactly
  k_out   <<<(B_ * 69 * S_ + 255) / 256, 256, 0, stream>>>(x1, fea1, bb1, Wb2, bb2, out);
}
